// VectorQuantizer_26551487824693
// MI455X (gfx1250) — matrix operand order repaired and checked by reading, not compiled, not run on hardware
//
#include <hip/hip_runtime.h>
#include <math.h>

typedef __attribute__((ext_vector_type(16))) _Float16 v16h;
typedef __attribute__((ext_vector_type(8)))  float    v8f;

#define K_CODES 4096
#define DDIM    64
#define NROWS   32768            // 8 * 4096
#define OFF_LOSS 2097152
#define OFF_IDX  2097153
#define OFF_PERP 2129921
#define KSLICES  4               // code-dim split per row-block (occupancy)
#define DIST_BIAS 512.0f         // makes all distances positive for key packing

// Workspace layout (bytes):
//   cb_h   : _Float16[4096*64]      @ 0        (524288)
//   cnorm  : float[4096]            @ 524288   (16384)
//   counts : uint[4096]             @ 540672   (16384)
//   loss   : float[1] (+pad)        @ 557056   (64)
//   keys   : u64[32768]             @ 557120   (262144)

#define WS_CNORM  524288
#define WS_COUNTS 540672
#define WS_LOSS   557056
#define WS_KEYS   557120

// ---------------------------------------------------------------------------
// Prep: f16 codebook copy, per-code squared norms, zero counts/loss,
//       init per-row argmin keys to +inf.
// ---------------------------------------------------------------------------
__global__ void vq_prep(const float* __restrict__ codebook,
                        _Float16* __restrict__ cb_h,
                        float* __restrict__ cnorm,
                        unsigned int* __restrict__ counts,
                        float* __restrict__ loss_acc,
                        unsigned long long* __restrict__ keys) {
    int k = blockIdx.x * blockDim.x + threadIdx.x;   // 0 .. 32767
    if (k < NROWS) keys[k] = 0xFFFFFFFFFFFFFFFFull;
    if (k == 0) *loss_acc = 0.f;
    if (k >= K_CODES) return;
    const float4* row = reinterpret_cast<const float4*>(codebook + (size_t)k * DDIM);
    uint2* dst = reinterpret_cast<uint2*>(cb_h + (size_t)k * DDIM);
    float nrm = 0.f;
#pragma unroll
    for (int i = 0; i < 16; ++i) {
        float4 f = row[i];
        nrm += f.x * f.x + f.y * f.y + f.z * f.z + f.w * f.w;
        union { _Float16 h[4]; uint2 u; } pk;
        pk.h[0] = (_Float16)f.x; pk.h[1] = (_Float16)f.y;
        pk.h[2] = (_Float16)f.z; pk.h[3] = (_Float16)f.w;
        dst[i] = pk.u;
    }
    cnorm[k]  = nrm;
    counts[k] = 0u;
}

// ---------------------------------------------------------------------------
// Search: WMMA distance partial-argmin. One wave owns 16 rows of z and a
// 1024-code slice (64 tiles of 16). Partials merge via u64 atomic-min on
// (biased-dist-bits << 32 | code) keys -- order-preserving, lowest-index ties.
// ---------------------------------------------------------------------------
__global__ void __launch_bounds__(256)
vq_search(const float* __restrict__ z,
          const _Float16* __restrict__ cb_h,
          const float* __restrict__ cnorm,
          unsigned long long* __restrict__ keys) {
    const int lane     = threadIdx.x & 31;
    const int gwave    = blockIdx.x * 8 + (threadIdx.x >> 5);
    const int rowbase  = (gwave >> 2) * 16;          // 16 rows per row-block
    const int tbase    = (gwave & (KSLICES - 1)) * (K_CODES / 16 / KSLICES);

    const int rA = lane & 15;        // A-matrix row / B-matrix column slot
    const int hi = lane >> 4;        // lane half

    // ---- Build A (16x64 f32 -> two 16x32 f16 WMMA A tiles), contiguous layout:
    // lane l holds row l%16, K in {b..b+15} with b = 16*(l>>4).
    v16h a0, a1;
    {
        const float4* zr =
            reinterpret_cast<const float4*>(z + (size_t)(rowbase + rA) * DDIM);
        const int b4 = hi * 4;
#pragma unroll
        for (int h = 0; h < 2; ++h) {
            float4 f0 = zr[h * 8 + b4 + 0];
            float4 f1 = zr[h * 8 + b4 + 1];
            float4 f2 = zr[h * 8 + b4 + 2];
            float4 f3 = zr[h * 8 + b4 + 3];
            v16h a;
            a[0]  = (_Float16)f0.x; a[1]  = (_Float16)f0.y;
            a[2]  = (_Float16)f0.z; a[3]  = (_Float16)f0.w;
            a[4]  = (_Float16)f1.x; a[5]  = (_Float16)f1.y;
            a[6]  = (_Float16)f1.z; a[7]  = (_Float16)f1.w;
            a[8]  = (_Float16)f2.x; a[9]  = (_Float16)f2.y;
            a[10] = (_Float16)f2.z; a[11] = (_Float16)f2.w;
            a[12] = (_Float16)f3.x; a[13] = (_Float16)f3.y;
            a[14] = (_Float16)f3.z; a[15] = (_Float16)f3.w;
            if (h == 0) a0 = a; else a1 = a;
        }
    }

    // ---- Running per-lane argmin: minv[v] covers row (v + 8*hi), column rA.
    float minv[8];
    int   mini[8];
#pragma unroll
    for (int i = 0; i < 8; ++i) { minv[i] = 3.4e38f; mini[i] = 0; }

    const int koff = hi * 16;        // B layout: half 0 holds K 0-15, half 1 K 16-31
    for (int t = tbase; t < tbase + K_CODES / 16 / KSLICES; ++t) {
        const int code = t * 16 + rA;
        const v16h b0v =
            *reinterpret_cast<const v16h*>(cb_h + (size_t)code * DDIM + koff);
        const v16h b1v =
            *reinterpret_cast<const v16h*>(cb_h + (size_t)code * DDIM + 32 + koff);
        const float cn = cnorm[code];

        v8f c = {};
        c = __builtin_amdgcn_wmma_f32_16x16x32_f16(false, a0, false, b0v,
                                                   (short)0, c, false, false);
        c = __builtin_amdgcn_wmma_f32_16x16x32_f16(false, a1, false, b1v,
                                                   (short)0, c, false, false);
#pragma unroll
        for (int v = 0; v < 8; ++v) {
            float d = cn - 2.0f * c[v];   // ||x||^2 omitted: constant per row
            if (d < minv[v]) { minv[v] = d; mini[v] = code; }
        }
    }

    // ---- Argmin reduction across the 16 column-slot lanes of each half.
#pragma unroll
    for (int v = 0; v < 8; ++v) {
#pragma unroll
        for (int off = 8; off >= 1; off >>= 1) {
            float ov = __shfl_xor(minv[v], off, 32);
            int   oi = __shfl_xor(mini[v], off, 32);
            if (ov < minv[v] || (ov == minv[v] && oi < mini[v])) {
                minv[v] = ov; mini[v] = oi;
            }
        }
    }

    // ---- Merge slice partials: u64 atomic-min on packed keys.
    if (rA == 0) {                   // lanes 0 and 16
        const int roff = hi * 8;
#pragma unroll
        for (int i = 0; i < 8; ++i) {
            union { float f; unsigned int u; } cv;
            cv.f = minv[i] + DIST_BIAS;          // strictly positive -> monotone bits
            unsigned long long key =
                ((unsigned long long)cv.u << 32) | (unsigned int)mini[i];
            atomicMin(&keys[rowbase + roff + i], key);
        }
    }
}

// ---------------------------------------------------------------------------
// Output: decode keys -> idx/counts, gather codebook (f32 exact) -> z_q (STE),
// commitment-loss partials. Two lanes per row.
// ---------------------------------------------------------------------------
__global__ void __launch_bounds__(256)
vq_output(const float* __restrict__ z,
          const float* __restrict__ codebook,
          const unsigned long long* __restrict__ keys,
          unsigned int* __restrict__ counts,
          float* __restrict__ loss_acc,
          float* __restrict__ out) {
    const int tid = blockIdx.x * 256 + threadIdx.x;  // 0 .. 65535
    const int row = tid >> 1;
    const int d0  = (tid & 1) * 32;

    const unsigned long long key = keys[row];
    const int code = (int)(key & 0xFFFFFFFFull);

    if ((tid & 1) == 0) {
        out[OFF_IDX + row] = (float)code;
        atomicAdd(&counts[code], 1u);
    }

    const float4* cb4 =
        reinterpret_cast<const float4*>(codebook + (size_t)code * DDIM + d0);
    const float4* z4 =
        reinterpret_cast<const float4*>(z + (size_t)row * DDIM + d0);
    float4* o4 = reinterpret_cast<float4*>(out + (size_t)row * DDIM + d0);

    float ls = 0.f;
#pragma unroll
    for (int i = 0; i < 8; ++i) {
        float4 cq = cb4[i];
        float4 zz = z4[i];
        o4[i] = cq;                  // z + sg(z_q - z) == z_q numerically
        float dx = zz.x - cq.x, dy = zz.y - cq.y;
        float dz = zz.z - cq.z, dw = zz.w - cq.w;
        ls += dx * dx + dy * dy + dz * dz + dw * dw;
    }
#pragma unroll
    for (int off = 16; off >= 1; off >>= 1) ls += __shfl_xor(ls, off, 32);
    if ((threadIdx.x & 31) == 0) atomicAdd(loss_acc, ls);
}

// ---------------------------------------------------------------------------
// Finalize: vq_loss and perplexity scalars.
// ---------------------------------------------------------------------------
__global__ void vq_finalize(const unsigned int* __restrict__ counts,
                            const float* __restrict__ loss_acc,
                            float* __restrict__ out) {
    __shared__ float sdata[256];
    const int tid = threadIdx.x;
    float s = 0.f;
    for (int k = tid; k < K_CODES; k += 256) {
        float p = (float)counts[k] * (1.0f / (float)NROWS);
        s += p * logf(p + 1e-10f);
    }
    sdata[tid] = s;
    __syncthreads();
    for (int w = 128; w > 0; w >>= 1) {
        if (tid < w) sdata[tid] += sdata[tid + w];
        __syncthreads();
    }
    if (tid == 0) {
        out[OFF_PERP] = expf(-sdata[0]);
        out[OFF_LOSS] = 0.25f * loss_acc[0] * (1.0f / (float)(NROWS * DDIM));
    }
}

// ---------------------------------------------------------------------------
extern "C" void kernel_launch(void* const* d_in, const int* in_sizes, int n_in,
                              void* d_out, int out_size, void* d_ws, size_t ws_size,
                              hipStream_t stream) {
    (void)in_sizes; (void)n_in; (void)out_size; (void)ws_size;
    const float* z        = (const float*)d_in[0];
    const float* codebook = (const float*)d_in[1];
    char* ws = (char*)d_ws;
    _Float16*           cb_h     = (_Float16*)(ws);
    float*              cnorm    = (float*)(ws + WS_CNORM);
    unsigned int*       counts   = (unsigned int*)(ws + WS_COUNTS);
    float*              loss_acc = (float*)(ws + WS_LOSS);
    unsigned long long* keys     = (unsigned long long*)(ws + WS_KEYS);
    float*              out      = (float*)d_out;

    vq_prep<<<NROWS / 256, 256, 0, stream>>>(codebook, cb_h, cnorm, counts,
                                             loss_acc, keys);
    // 2048 row-blocks x 4 code slices = 8192 waves; 8 waves per block.
    vq_search<<<8192 / 8, 256, 0, stream>>>(z, cb_h, cnorm, keys);
    vq_output<<<(NROWS * 2) / 256, 256, 0, stream>>>(z, codebook, keys, counts,
                                                     loss_acc, out);
    vq_finalize<<<1, 256, 0, stream>>>(counts, loss_acc, out);
}
